// ResidualVectorQuantizer_10307921510620
// MI455X (gfx1250) — matrix operand order repaired and checked by reading, not compiled, not run on hardware
//
#include <hip/hip_runtime.h>
#include <hip/hip_bf16.h>
#include <stdint.h>

// ---------------------------------------------------------------------------
// Residual VQ for MI455X (gfx1250): BF16 WMMA distance GEMM + TDM staging.
// N=131072 rows, D=256, K=1024 codes, 4 stages fused in one kernel.
// Round 2: 4-deep TDM/LDS pipeline (1 barrier per tile), 2-deep register
// prefetch of B fragments to overlap ds_load latency with the WMMA pipe.
// ---------------------------------------------------------------------------

typedef __bf16        v16bf __attribute__((ext_vector_type(16)));
typedef float         v8f   __attribute__((ext_vector_type(8)));
typedef unsigned int  v8u   __attribute__((ext_vector_type(8)));
typedef unsigned int  u32x4 __attribute__((ext_vector_type(4)));
typedef int           i32x8 __attribute__((ext_vector_type(8)));
typedef int           i32x4 __attribute__((ext_vector_type(4)));

#define RVQ_NQ   4
#define RVQ_K    1024
#define RVQ_D    256
#define RVQ_N    (32 * 4096)
#define KTILES   64            // K / 16 codes per tile
#define DTILES   8             // D / 32 (wmma K-dim)
#define TILE_BYTES 8192        // one ktile of swizzled bf16 B-fragments
#define NBUF     4             // LDS pipeline depth

// ---------------------------------------------------------------------------
// TDM: async-load one 8KB codebook fragment tile (contiguous) into LDS.
// Descriptor packed per CDNA5 ISA ch.8 (group0 128b, group1 256b), 1-D tile:
// data_size=4B, tile_dim0 = tensor_dim0 = stride0 = 2048 dwords.
// ---------------------------------------------------------------------------
__device__ __forceinline__ void tdm_load_tile(uint32_t lds_off, uint64_t gaddr) {
  u32x4 g0;
  g0.x = 1u;                                              // count=1 (valid), no gather
  g0.y = lds_off;                                         // lds_addr
  g0.z = (uint32_t)(gaddr & 0xFFFFFFFFull);               // global_addr[31:0]
  g0.w = (uint32_t)((gaddr >> 32) & 0x1FFFFFFull)         // global_addr[56:32]
       | (2u << 30);                                      // type=2 ("image")
  i32x8 g1;
  g1[0] = (int)(2u << 16);          // workgroup_mask=0, data_size=2 (4B)
  g1[1] = (int)(2048u << 16);       // tensor_dim0[15:0] << 16
  g1[2] = (int)(1u << 16);          // tensor_dim0[31:16]=0 | tensor_dim1=1
  g1[3] = (int)(2048u << 16);       // tile_dim0 = 2048
  g1[4] = 0;                        // tile_dim1=0 (1-D), tile_dim2=0
  g1[5] = 2048;                     // tensor_dim0_stride[31:0]
  g1[6] = 0;
  g1[7] = 0;
  i32x4 z4 = {0, 0, 0, 0};
#if defined(__clang_major__) && (__clang_major__ >= 23)
  i32x8 z8 = {0, 0, 0, 0, 0, 0, 0, 0};
  __builtin_amdgcn_tensor_load_to_lds(g0, g1, z4, z4, z8, 0);
#else
  __builtin_amdgcn_tensor_load_to_lds(g0, g1, z4, z4, 0);
#endif
}

// ---------------------------------------------------------------------------
// Prepass 1: swizzle fp32 codebooks into bf16 B-fragment layout.
// B-frag (16-bit, 32x16 KxN): lane l holds code (N) l%16; its 16 halves are
// d0+8h+{0..7}, d0+16+8h+{0..7} (h=l/16). One thread builds one lane's 8 dwords.
// Layout in ws: [stage][ktile][dtile][lane][8 dwords]  (8KB per (stage,ktile))
// ---------------------------------------------------------------------------
__global__ void rvq_swizzle_codebook(const float* __restrict__ cb,
                                     uint32_t* __restrict__ wsB) {
  const int gid   = blockIdx.x * 256 + threadIdx.x;   // 0 .. 65535
  const int lane  = gid & 31;
  const int dtile = (gid >> 5) & 7;
  const int ktile = (gid >> 8) & 63;
  const int stage = gid >> 14;
  const float* base =
      cb + (size_t)(stage * RVQ_K + ktile * 16 + (lane & 15)) * RVQ_D + dtile * 32 + (lane >> 4) * 8;
  v16bf h;
#pragma unroll
  for (int j = 0; j < 16; ++j) h[j] = (__bf16)base[j < 8 ? j : j + 8];
  *(v8u*)(wsB + (size_t)gid * 8) = __builtin_bit_cast(v8u, h);
}

// ---------------------------------------------------------------------------
// Prepass 2: ||c||^2 per (stage, code). One wave per code.
// ---------------------------------------------------------------------------
__global__ void rvq_codebook_norms(const float* __restrict__ cb,
                                   float* __restrict__ wsC) {
  const int wid  = blockIdx.x * 8 + (threadIdx.x >> 5);  // 0 .. 4095
  const int lane = threadIdx.x & 31;
  const float* p = cb + (size_t)wid * RVQ_D + lane * 8;
  float s = 0.f;
#pragma unroll
  for (int j = 0; j < 8; ++j) s += p[j] * p[j];
#pragma unroll
  for (int m = 1; m < 32; m <<= 1) s += __shfl_xor(s, m, 32);
  if (lane == 0) wsC[wid] = s;
}

// ---------------------------------------------------------------------------
// Main kernel: 8 waves/block, 16 rows/wave, residual resident in VGPRs.
// Residual layout per lane matches the A-fragment layout:
//   lane l (<16) : row l,   d in {d0+0..7,  d0+16..23} per dtile
//   lane l (>=16): row l-16, d in {d0+8..15, d0+24..31} per dtile
// rres[t*16 + j]: j<8 -> d0+delta+j ; j>=8 -> d0+16+delta+(j-8)
// ---------------------------------------------------------------------------
#define SET4(b, v)  { rres[(b)] = (v).x;  rres[(b)+1] = (v).y;  rres[(b)+2] = (v).z;  rres[(b)+3] = (v).w; }
#define SUB4(b, v)  { rres[(b)] -= (v).x; rres[(b)+1] -= (v).y; rres[(b)+2] -= (v).z; rres[(b)+3] -= (v).w; }
#define OUT4(b, v)  make_float4((v).x - rres[(b)], (v).y - rres[(b)+1], (v).z - rres[(b)+2], (v).w - rres[(b)+3])

__global__ __launch_bounds__(256) void rvq_main(
    const float* __restrict__ x, const float* __restrict__ cb,
    const uint32_t* __restrict__ wsB, const float* __restrict__ wsC,
    float* __restrict__ outQ, float* __restrict__ outCodes) {
  __shared__ __align__(32) uint32_t lds_b[NBUF][2048];  // 4-deep B-tile pipeline
  __shared__ float lds_cn[RVQ_K];                       // code norms for stage
  __shared__ int   lds_arg[8][16];                      // per-wave winning codes

  const int tid     = threadIdx.x;
  const int wid     = tid >> 5;
  const int lane    = tid & 31;
  const int halfsel = lane >> 4;     // 0 | 1
  const int delta   = halfsel * 8;
  const int r       = lane & 15;     // row served by this lane (within wave)
  const int row     = blockIdx.x * 128 + wid * 16 + r;
  const float* xrow = x + (size_t)row * RVQ_D;
  const uint64_t wsB_base = (uint64_t)(uintptr_t)wsB;

  // ---- load x into the residual register file -----------------------------
  float rres[128];
#pragma unroll
  for (int t = 0; t < DTILES; ++t) {
    const float* p = xrow + t * 32 + delta;
    float4 a0 = *(const float4*)(p);
    float4 a1 = *(const float4*)(p + 4);
    float4 b0 = *(const float4*)(p + 16);
    float4 b1 = *(const float4*)(p + 20);
    SET4(t * 16 + 0, a0); SET4(t * 16 + 4, a1);
    SET4(t * 16 + 8, b0); SET4(t * 16 + 12, b1);
  }

  // ---- TDM pipeline prologue: tiles 0..2 into buffers 0..2 ----------------
  if (wid == 0) {
#pragma unroll
    for (int i = 0; i < NBUF - 1; ++i)
      tdm_load_tile((uint32_t)(uintptr_t)&lds_b[i][0],
                    wsB_base + (uint64_t)i * TILE_BYTES);
  }

  for (int stage = 0; stage < RVQ_NQ; ++stage) {
    // stage the code norms (consumed only after the next start-barrier)
#pragma unroll
    for (int i = 0; i < 4; ++i)
      lds_cn[tid + i * 256] = wsC[stage * RVQ_K + tid + i * 256];

    // build bf16 A fragments from the fp32 residual (lane-local convert)
    v16bf abf[DTILES];
#pragma unroll
    for (int t = 0; t < DTILES; ++t) {
      v16bf a;
#pragma unroll
      for (int j = 0; j < 16; ++j) a[j] = (__bf16)rres[t * 16 + j];
      abf[t] = a;
    }

    float minv[8];
    int   mini[8];
#pragma unroll
    for (int v = 0; v < 8; ++v) { minv[v] = 3.4e38f; mini[v] = 0; }

    for (int k = 0; k < KTILES; ++k) {
      // issue tile g+3 (wraps mod 256 at the tail: harmless refetch) and
      // guarantee tile g has landed; buffer for tile g is g&3 == k&3.
      if (wid == 0) {
        const int gnext = (stage * KTILES + k + NBUF - 1) & 255;
        tdm_load_tile((uint32_t)(uintptr_t)&lds_b[gnext & (NBUF - 1)][0],
                      wsB_base + (uint64_t)gnext * TILE_BYTES);
        __builtin_amdgcn_s_wait_tensorcnt(NBUF - 1);
      }
      __syncthreads();  // tile k visible; buffer (k+3)&3 free for rewrite

      // 16x16 distances vs 16 codes: 8 chained WMMAs over D=256,
      // with a 2-deep register prefetch of B fragments.
      const uint32_t* bbuf = &lds_b[k & (NBUF - 1)][0];
      v8u cur = *(const v8u*)(bbuf + lane * 8);
      v8f acc = {};
#pragma unroll
      for (int t = 0; t < DTILES; ++t) {
        v8u nxt = cur;
        if (t + 1 < DTILES) nxt = *(const v8u*)(bbuf + ((t + 1) * 32 + lane) * 8);
        acc = __builtin_amdgcn_wmma_f32_16x16x32_bf16(
            false, abf[t], false, __builtin_bit_cast(v16bf, cur),
            (short)0, acc, false, false);
        cur = nxt;
      }

      // dist = ||c||^2 - 2 r.c  (||r||^2 constant per row -> dropped)
      const float cn   = lds_cn[k * 16 + r];
      const int   code = k * 16 + r;
#pragma unroll
      for (int v = 0; v < 8; ++v) {
        float d = cn - 2.0f * acc[v];
        if (d < minv[v]) { minv[v] = d; mini[v] = code; }
      }
    }

    // reduce argmin across the 16 column-lanes of each half-wave
#pragma unroll
    for (int m = 1; m <= 8; m <<= 1) {
#pragma unroll
      for (int v = 0; v < 8; ++v) {
        float ov = __shfl_xor(minv[v], m, 32);
        int   oi = __shfl_xor(mini[v], m, 32);
        if (ov < minv[v]) { minv[v] = ov; mini[v] = oi; }
      }
    }
    // lanes 0 / 16 hold rows 0..7 / 8..15 -> publish per-wave winners
    if ((lane & 15) == 0) {
#pragma unroll
      for (int v = 0; v < 8; ++v) lds_arg[wid][halfsel * 8 + v] = mini[v];
    }
    __syncthreads();  // also orders: all k-loop LDS reads done before reuse

    const int code = lds_arg[wid][r];
    if (lane < 16) outCodes[(size_t)row * RVQ_NQ + stage] = (float)code;

    // gather chosen code (fp32) and update residual: r -= c
    const float* cvec = cb + (size_t)(stage * RVQ_K + code) * RVQ_D;
#pragma unroll
    for (int t = 0; t < DTILES; ++t) {
      const float* p = cvec + t * 32 + delta;
      float4 a0 = *(const float4*)(p);
      float4 a1 = *(const float4*)(p + 4);
      float4 b0 = *(const float4*)(p + 16);
      float4 b1 = *(const float4*)(p + 20);
      SUB4(t * 16 + 0, a0); SUB4(t * 16 + 4, a1);
      SUB4(t * 16 + 8, b0); SUB4(t * 16 + 12, b1);
    }
  }

  // quantised = x - residual_final
  float* orow = outQ + (size_t)row * RVQ_D;
#pragma unroll
  for (int t = 0; t < DTILES; ++t) {
    const float* p = xrow + t * 32 + delta;
    float4 a0 = *(const float4*)(p);
    float4 a1 = *(const float4*)(p + 4);
    float4 b0 = *(const float4*)(p + 16);
    float4 b1 = *(const float4*)(p + 20);
    *(float4*)(orow + t * 32 + delta)      = OUT4(t * 16 + 0, a0);
    *(float4*)(orow + t * 32 + delta + 4)  = OUT4(t * 16 + 4, a1);
    *(float4*)(orow + t * 32 + delta + 16) = OUT4(t * 16 + 8, b0);
    *(float4*)(orow + t * 32 + delta + 20) = OUT4(t * 16 + 12, b1);
  }
}

// ---------------------------------------------------------------------------
extern "C" void kernel_launch(void* const* d_in, const int* in_sizes, int n_in,
                              void* d_out, int out_size, void* d_ws, size_t ws_size,
                              hipStream_t stream) {
  (void)in_sizes; (void)n_in; (void)out_size; (void)ws_size;
  const float* x  = (const float*)d_in[0];    // (B,T,D) fp32
  const float* cb = (const float*)d_in[1];    // (NUM_Q,K,D) fp32

  uint32_t* wsB = (uint32_t*)d_ws;                                  // 2 MB swizzled bf16 frags
  float*    wsC = (float*)((char*)d_ws + (size_t)RVQ_NQ * KTILES * TILE_BYTES); // 16 KB norms

  float* outQ     = (float*)d_out;                  // quantised (B*T*D)
  float* outCodes = outQ + (size_t)RVQ_N * RVQ_D;   // codes (B*T*NUM_Q) as float

  rvq_swizzle_codebook<<<256, 256, 0, stream>>>(cb, wsB);
  rvq_codebook_norms<<<512, 256, 0, stream>>>(cb, wsC);
  rvq_main<<<1024, 256, 0, stream>>>(x, cb, wsB, wsC, outQ, outCodes);
}
